// SA_Layer_8211977470432
// MI455X (gfx1250) — matrix operand order repaired and checked by reading, not compiled, not run on hardware
//
#include <hip/hip_runtime.h>
#include <hip/hip_bf16.h>

// ---------------------------------------------------------------------------
// SA_Layer for MI455X (gfx1250, wave32, WMMA).
// B=4, C=256, N=4096.  All big GEMMs use v_wmma_f32_16x16x32_f16 (f16 in,
// f32 accumulate); softmax / BN statistics stay in f32.
// ---------------------------------------------------------------------------

typedef _Float16 f16;
typedef __attribute__((ext_vector_type(16))) _Float16 v16h;
typedef __attribute__((ext_vector_type(8)))  _Float16 v8h;
typedef __attribute__((ext_vector_type(8)))  float    v8f;

constexpr int B  = 4;
constexpr int C  = 256;
constexpr int C4 = 64;
constexpr int N  = 4096;
constexpr int NT = 16;          // n-chunk per projection block

// ---- WMMA wrapper ---------------------------------------------------------
__device__ __forceinline__ v8f wmma_f16(v16h a, v16h b, v8f c) {
    // 8 args: (neg_a, A, neg_b, B, c_mod, C, reuse_a, reuse_b)
    return __builtin_amdgcn_wmma_f32_16x16x32_f16(false, a, false, b,
                                                  (short)0, c, false, false);
}

// ---- Fragment loaders (ISA 7.12.2 layouts, wave32) ------------------------
// A (16x32 f16, row-major source, row stride ldk):
//   lane<16 : row=lane,    K = {0..7, 16..23}
//   lane>=16: row=lane-16, K = {8..15, 24..31}
__device__ __forceinline__ v16h ld_a_frag(const f16* __restrict__ base, int ldk) {
    const int lane = threadIdx.x & 31;
    const int half = lane >> 4;
    const int r    = lane & 15;
    const f16* p = base + (size_t)r * ldk + half * 8;
    v8h lo = *(const v8h*)(p);
    v8h hi = *(const v8h*)(p + 16);
    v16h a;
#pragma unroll
    for (int i = 0; i < 8; ++i) { a[i] = lo[i]; a[8 + i] = hi[i]; }
    return a;
}

// B (32x16 f16). Source stores columns as rows (row = output column, K
// contiguous): lane<16 holds K={0..7,16..23} of column lane; lane>=16 holds K={8..15,24..31}.
__device__ __forceinline__ v16h ld_b_frag(const f16* __restrict__ base, int ldk) {
    const int lane = threadIdx.x & 31;
    const int half = lane >> 4;
    const int col  = lane & 15;
    const f16* p = base + (size_t)col * ldk + half * 8;
    v8h lo = *(const v8h*)(p);
    v8h hi = *(const v8h*)(p + 16);
    v16h b;
#pragma unroll
    for (int i = 0; i < 8; ++i) { b[i] = lo[i]; b[8 + i] = hi[i]; }
    return b;
}

// ---- half-wave (16 lane) reductions (used ONCE per pass, not per tile) ----
__device__ __forceinline__ float half_reduce_max(float x) {
#pragma unroll
    for (int m = 1; m < 16; m <<= 1) x = fmaxf(x, __shfl_xor(x, m, 32));
    return x;
}
__device__ __forceinline__ float half_reduce_sum(float x) {
#pragma unroll
    for (int m = 1; m < 16; m <<= 1) x += __shfl_xor(x, m, 32);
    return x;
}

// ===========================================================================
// K0: Wt -> f16
// ===========================================================================
__global__ void cvt_wt_kernel(const float* __restrict__ Wt, f16* __restrict__ Wth) {
    int i = blockIdx.x * 256 + threadIdx.x;
    if (i < C * C) Wth[i] = (f16)Wt[i];
}

// ===========================================================================
// K1: projections.  q_h[b][n][o], k_h[b][m][o] (= x_k transposed), v_h[b][c][n]
// ===========================================================================
__global__ void proj_kernel(const float* __restrict__ x,
                            const float* __restrict__ Wq,
                            const float* __restrict__ Wk,
                            const float* __restrict__ Wv,
                            const float* __restrict__ bv,
                            f16* __restrict__ qh, f16* __restrict__ kh,
                            f16* __restrict__ vh) {
    const int b  = blockIdx.y;
    const int n0 = blockIdx.x * NT;
    const int t  = threadIdx.x;

    __shared__ float xs[C][NT];   // 16 KB
    const float* xb = x + (size_t)b * C * N;
    for (int i = t; i < C * NT; i += 256) {
        int c = i / NT, j = i % NT;
        xs[c][j] = xb[(size_t)c * N + n0 + j];
    }
    __syncthreads();

    float vacc[NT], qacc[NT];
#pragma unroll
    for (int j = 0; j < NT; ++j) { vacc[j] = 0.f; qacc[j] = 0.f; }

    const float* wv  = Wv + (size_t)t * C;
    const float* wqk = (t < 64) ? (Wq + (size_t)t * C)
                                : ((t < 128) ? (Wk + (size_t)(t - 64) * C)
                                             : (Wq + (size_t)(t & 63) * C));
    for (int c = 0; c < C; ++c) {
        float wvc = wv[c];
        float wq  = wqk[c];
#pragma unroll
        for (int j = 0; j < NT; ++j) {
            float xcj = xs[c][j];
            vacc[j] = fmaf(wvc, xcj, vacc[j]);
            qacc[j] = fmaf(wq,  xcj, qacc[j]);
        }
    }

    float bvt = bv[t];
    v16h hv;
#pragma unroll
    for (int j = 0; j < NT; ++j) hv[j] = (f16)(vacc[j] + bvt);
    *(v16h*)(vh + ((size_t)b * C + t) * N + n0) = hv;

    if (t < 64) {
        for (int j = 0; j < NT; ++j)
            qh[((size_t)b * N + n0 + j) * C4 + t] = (f16)qacc[j];
    } else if (t < 128) {
        for (int j = 0; j < NT; ++j)
            kh[((size_t)b * N + n0 + j) * C4 + (t - 64)] = (f16)qacc[j];
    }
}

// ===========================================================================
// K2: energy + row softmax + column sums.  One block = 16 query rows, 8 waves.
// 3 GEMM passes (rowmax / rowsum / store).  Each pass iteration processes TWO
// independent 16x16 m-tiles (independent WMMA chains fill the WMMA->VALU
// hazard slots).  Per-lane partials accumulate in registers; cross-lane
// reduction happens once per pass.  Pt[b][m][n] (f16), colsum[b][m].
// ===========================================================================
__global__ void attn_softmax_kernel(const f16* __restrict__ qh,
                                    const f16* __restrict__ kh,
                                    f16* __restrict__ Pt,
                                    float* __restrict__ colsum) {
    const int b    = blockIdx.x;
    const int n0   = blockIdx.y * 16;
    const int wave = threadIdx.x >> 5;
    const int lane = threadIdx.x & 31;
    const int half = lane >> 4;
    const int col  = lane & 15;

    __shared__ float red[8][16];
    __shared__ float srmax[16];
    __shared__ float srsum[16];

    const f16* qbase = qh + ((size_t)b * N + n0) * C4;
    v16h a0 = ld_a_frag(qbase,      C4);
    v16h a1 = ld_a_frag(qbase + 32, C4);
    const f16* kb = kh + (size_t)b * N * C4;

    // ---- pass 1: row max (two tiles / iter, VALU only in loop) ----
    float rmax[8];
#pragma unroll
    for (int v = 0; v < 8; ++v) rmax[v] = -3.0e38f;
    for (int p = wave; p < N / 32; p += 8) {
        const f16* k0 = kb + (size_t)p * 32 * C4;
        const f16* k1 = k0 + 16 * C4;
        v8f c0 = {}, c1 = {};
        c0 = wmma_f16(a0, ld_b_frag(k0,      C4), c0);
        c1 = wmma_f16(a0, ld_b_frag(k1,      C4), c1);
        c0 = wmma_f16(a1, ld_b_frag(k0 + 32, C4), c0);
        c1 = wmma_f16(a1, ld_b_frag(k1 + 32, C4), c1);
#pragma unroll
        for (int v = 0; v < 8; ++v)
            rmax[v] = fmaxf(rmax[v], fmaxf(c0[v], c1[v]));
    }
#pragma unroll
    for (int v = 0; v < 8; ++v) rmax[v] = half_reduce_max(rmax[v]);
    if (col == 0) {
#pragma unroll
        for (int v = 0; v < 8; ++v) red[wave][half * 8 + v] = rmax[v];
    }
    __syncthreads();
    if (threadIdx.x < 16) {
        float m = -3.0e38f;
        for (int w = 0; w < 8; ++w) m = fmaxf(m, red[w][threadIdx.x]);
        srmax[threadIdx.x] = m;
    }
    __syncthreads();
    float rmx[8];
#pragma unroll
    for (int v = 0; v < 8; ++v) rmx[v] = srmax[half * 8 + v];

    // ---- pass 2: row sum of exp (two tiles / iter) ----
    float rs[8];
#pragma unroll
    for (int v = 0; v < 8; ++v) rs[v] = 0.f;
    for (int p = wave; p < N / 32; p += 8) {
        const f16* k0 = kb + (size_t)p * 32 * C4;
        const f16* k1 = k0 + 16 * C4;
        v8f c0 = {}, c1 = {};
        c0 = wmma_f16(a0, ld_b_frag(k0,      C4), c0);
        c1 = wmma_f16(a0, ld_b_frag(k1,      C4), c1);
        c0 = wmma_f16(a1, ld_b_frag(k0 + 32, C4), c0);
        c1 = wmma_f16(a1, ld_b_frag(k1 + 32, C4), c1);
#pragma unroll
        for (int v = 0; v < 8; ++v)
            rs[v] += __expf(c0[v] - rmx[v]) + __expf(c1[v] - rmx[v]);
    }
#pragma unroll
    for (int v = 0; v < 8; ++v) rs[v] = half_reduce_sum(rs[v]);
    if (col == 0) {
#pragma unroll
        for (int v = 0; v < 8; ++v) red[wave][half * 8 + v] = rs[v];
    }
    __syncthreads();
    if (threadIdx.x < 16) {
        float s = 0.f;
        for (int w = 0; w < 8; ++w) s += red[w][threadIdx.x];
        srsum[threadIdx.x] = s;
    }
    __syncthreads();
    float rinv[8];
#pragma unroll
    for (int v = 0; v < 8; ++v) rinv[v] = 1.0f / srsum[half * 8 + v];

    // ---- pass 3: normalized P (stored transposed) + column sums ----
    for (int p = wave; p < N / 32; p += 8) {
        const int m0 = p * 32;
        const f16* k0 = kb + (size_t)m0 * C4;
        const f16* k1 = k0 + 16 * C4;
        v8f c0 = {}, c1 = {};
        c0 = wmma_f16(a0, ld_b_frag(k0,      C4), c0);
        c1 = wmma_f16(a0, ld_b_frag(k1,      C4), c1);
        c0 = wmma_f16(a1, ld_b_frag(k0 + 32, C4), c0);
        c1 = wmma_f16(a1, ld_b_frag(k1 + 32, C4), c1);

        float cs0 = 0.f, cs1 = 0.f;
        v8h hv0, hv1;
#pragma unroll
        for (int v = 0; v < 8; ++v) {
            float p0 = __expf(c0[v] - rmx[v]) * rinv[v];
            float p1 = __expf(c1[v] - rmx[v]) * rinv[v];
            cs0 += p0; cs1 += p1;
            hv0[v] = (f16)p0; hv1[v] = (f16)p1;
        }
        // rows v..v+7 of this half are consecutive n -> one 16B store each
        f16* prow = Pt + ((size_t)b * N + m0 + col) * N + n0 + half * 8;
        *(v8h*)(prow)                     = hv0;
        *(v8h*)(prow + (size_t)16 * N)    = hv1;
        cs0 += __shfl_xor(cs0, 16, 32);   // add other half's 8 rows
        cs1 += __shfl_xor(cs1, 16, 32);
        if (half == 0) {
            atomicAdd(&colsum[b * N + m0 + col],      cs0);
            atomicAdd(&colsum[b * N + m0 + 16 + col], cs1);
        }
    }
}

// ===========================================================================
// K3: x_r = (v · P) / (1e-9 + colsum);  d = x - x_r  (stored f16, [b][m][c])
// Two independent accumulator chains over K for WMMA ILP.
// ===========================================================================
__global__ void xr_d_kernel(const f16* __restrict__ vh,
                            const f16* __restrict__ Pt,
                            const float* __restrict__ colsum,
                            const float* __restrict__ x,
                            f16* __restrict__ dh) {
    const int b    = blockIdx.z;
    const int c0   = blockIdx.y * 16;
    const int wave = threadIdx.x >> 5;
    const int m0   = (blockIdx.x * 4 + wave) * 16;
    const int lane = threadIdx.x & 31;
    const int half = lane >> 4;
    const int col  = lane & 15;

    const f16* abase = vh + ((size_t)b * C + c0) * N;   // rows c, K = n
    const f16* bbase = Pt + ((size_t)b * N + m0) * N;   // rows m, K = n
    v8f acc0 = {}, acc1 = {};
    for (int k0 = 0; k0 < N; k0 += 64) {
        acc0 = wmma_f16(ld_a_frag(abase + k0,      N), ld_b_frag(bbase + k0,      N), acc0);
        acc1 = wmma_f16(ld_a_frag(abase + k0 + 32, N), ld_b_frag(bbase + k0 + 32, N), acc1);
    }

    float cinv = 1.0f / (1e-9f + colsum[b * N + m0 + col]);
    v8h hv;
#pragma unroll
    for (int v = 0; v < 8; ++v) {
        int c = c0 + half * 8 + v;
        float xr = (acc0[v] + acc1[v]) * cinv;
        hv[v] = (f16)(x[((size_t)b * C + c) * N + m0 + col] - xr);
    }
    *(v8h*)(dh + ((size_t)b * N + m0 + col) * C + c0 + half * 8) = hv;
}

// ===========================================================================
// K4: t = Wt · d + bt  (stored f32 [b][m][o]) + BN partial sums per channel
// ===========================================================================
__global__ void trans_bn_kernel(const f16* __restrict__ Wth,
                                const f16* __restrict__ dh,
                                const float* __restrict__ bt,
                                float* __restrict__ t,
                                float* __restrict__ bnsum,
                                float* __restrict__ bnsumsq) {
    const int b    = blockIdx.z;
    const int o0   = blockIdx.y * 16;
    const int wave = threadIdx.x >> 5;
    const int m0   = (blockIdx.x * 8 + wave) * 16;
    const int lane = threadIdx.x & 31;
    const int half = lane >> 4;
    const int col  = lane & 15;

    __shared__ float ssum[16], ssq[16];
    if (threadIdx.x < 16) { ssum[threadIdx.x] = 0.f; ssq[threadIdx.x] = 0.f; }
    __syncthreads();

    v16h a[8];
#pragma unroll
    for (int kk = 0; kk < 8; ++kk)
        a[kk] = ld_a_frag(Wth + (size_t)o0 * C + kk * 32, C);

    const f16* bbase = dh + ((size_t)b * N + m0) * C;   // rows m, K = c
    v8f acc0 = {}, acc1 = {};
#pragma unroll
    for (int kk = 0; kk < 8; kk += 2) {
        acc0 = wmma_f16(a[kk],     ld_b_frag(bbase + kk * 32,      C), acc0);
        acc1 = wmma_f16(a[kk + 1], ld_b_frag(bbase + kk * 32 + 32, C), acc1);
    }

    v8f tv;
#pragma unroll
    for (int v = 0; v < 8; ++v) tv[v] = acc0[v] + acc1[v] + bt[o0 + half * 8 + v];
    *(v8f*)(t + ((size_t)b * N + m0 + col) * C + o0 + half * 8) = tv;

#pragma unroll
    for (int v = 0; v < 8; ++v) {
        float rsum = half_reduce_sum(tv[v]);
        float rsq  = half_reduce_sum(tv[v] * tv[v]);
        if (col == 0) {
            atomicAdd(&ssum[half * 8 + v], rsum);
            atomicAdd(&ssq [half * 8 + v], rsq);
        }
    }
    __syncthreads();
    if (threadIdx.x < 16) {
        atomicAdd(&bnsum  [o0 + threadIdx.x], ssum[threadIdx.x]);
        atomicAdd(&bnsumsq[o0 + threadIdx.x], ssq [threadIdx.x]);
    }
}

// ===========================================================================
// K5: out = x + relu(BN(t) * gamma + beta)   (LDS transpose [m][o] -> [c][n])
// ===========================================================================
__global__ void bn_out_kernel(const float* __restrict__ t,
                              const float* __restrict__ x,
                              const float* __restrict__ gamma,
                              const float* __restrict__ beta,
                              const float* __restrict__ bnsum,
                              const float* __restrict__ bnsumsq,
                              float* __restrict__ out) {
    __shared__ float tile[32][33];
    const int b  = blockIdx.z;
    const int o0 = blockIdx.y * 32;
    const int m0 = blockIdx.x * 32;
    const float inv = 1.0f / (float)(B * N);

    const float* tb = t + (size_t)b * N * C;
    for (int i = threadIdx.x; i < 32 * 32; i += 256) {
        int j  = i >> 5;    // m offset
        int ii = i & 31;    // o offset
        tile[j][ii] = tb[(size_t)(m0 + j) * C + o0 + ii];
    }
    __syncthreads();
    for (int i = threadIdx.x; i < 32 * 32; i += 256) {
        int io = i >> 5;    // o offset
        int j  = i & 31;    // m offset
        int c = o0 + io, n = m0 + j;
        float mean = bnsum[c] * inv;
        float var  = bnsumsq[c] * inv - mean * mean;
        float th   = (tile[j][io] - mean) * rsqrtf(var + 1e-5f);
        float val  = fmaxf(th * gamma[c] + beta[c], 0.f);
        size_t idx = ((size_t)b * C + c) * N + n;
        out[idx] = x[idx] + val;
    }
}

// ===========================================================================
extern "C" void kernel_launch(void* const* d_in, const int* in_sizes, int n_in,
                              void* d_out, int out_size, void* d_ws, size_t ws_size,
                              hipStream_t stream) {
    (void)in_sizes; (void)n_in; (void)out_size; (void)ws_size;
    const float* x     = (const float*)d_in[0];
    const float* Wq    = (const float*)d_in[1];
    const float* Wk    = (const float*)d_in[2];
    const float* Wv    = (const float*)d_in[3];
    const float* bv    = (const float*)d_in[4];
    const float* Wt    = (const float*)d_in[5];
    const float* bt    = (const float*)d_in[6];
    const float* gamma = (const float*)d_in[7];
    const float* beta  = (const float*)d_in[8];
    float* out = (float*)d_out;

    char* w = (char*)d_ws;
    auto alloc = [&](size_t bytes) -> char* {
        char* p = w;
        w += (bytes + 255) & ~(size_t)255;
        return p;
    };
    f16*   qh     = (f16*)  alloc((size_t)B * N * C4 * 2);          //   2 MB
    f16*   kh     = (f16*)  alloc((size_t)B * N * C4 * 2);          //   2 MB
    f16*   vh     = (f16*)  alloc((size_t)B * C * N  * 2);          //   8 MB
    f16*   dh     = (f16*)  alloc((size_t)B * C * N  * 2);          //   8 MB
    float* tbuf   = (float*)alloc((size_t)B * N * C  * 4);          //  16 MB
    f16*   Wth    = (f16*)  alloc((size_t)C * C * 2);               // 128 KB
    float* colsum = (float*)alloc(((size_t)B * N + 2 * C) * 4);     //  66 KB
    float* bnsum  = colsum + (size_t)B * N;
    float* bnsumq = bnsum + C;
    f16*   Pt     = (f16*)  alloc((size_t)B * N * N * 2);           // 128 MB

    hipMemsetAsync(colsum, 0, ((size_t)B * N + 2 * C) * 4, stream);

    cvt_wt_kernel   <<<C * C / 256, 256, 0, stream>>>(Wt, Wth);
    proj_kernel     <<<dim3(N / NT, B), 256, 0, stream>>>(x, Wq, Wk, Wv, bv, qh, kh, vh);
    attn_softmax_kernel<<<dim3(B, N / 16), 256, 0, stream>>>(qh, kh, Pt, colsum);
    xr_d_kernel     <<<dim3(N / 64, C / 16, B), 128, 0, stream>>>(vh, Pt, colsum, x, dh);
    trans_bn_kernel <<<dim3(N / 128, C / 16, B), 256, 0, stream>>>(Wth, dh, bt, tbuf, bnsum, bnsumq);
    bn_out_kernel   <<<dim3(N / 32, C / 32, B), 256, 0, stream>>>(tbuf, x, gamma, beta, bnsum, bnsumq, out);
}
